// NeighConv_37649683316960
// MI455X (gfx1250) — matrix operand order repaired and checked by reading, not compiled, not run on hardware
//
#include <hip/hip_runtime.h>

#define CB 2
#define CCH 256
#define NP 4096
#define KNN 16
#define MTOT (CB * NP)            // 8192 points total
#define JSPLIT 8                  // candidate-range split per query tile
#define JT_PER (NP / 16 / JSPLIT) // 32 candidate tiles per wave

typedef __attribute__((ext_vector_type(16))) __bf16        v16bf;
typedef __attribute__((ext_vector_type(8)))  float         v8f;
typedef __attribute__((ext_vector_type(4)))  unsigned int  v4u;

union FragU { struct { v4u lo, hi; } p; v16bf v; };
union Pack8 { v4u u; __bf16 h[8]; };

// A fragment (16x32 bf16, MxK): lane half h=0 -> K[k0..k0+7] + K[k0+16..k0+23],
// h=1 -> K[k0+8..k0+15] + K[k0+24..k0+31]; rowk = row base + k0.
__device__ __forceinline__ v16bf load_frag_a(const __bf16* rowk, int h) {
  FragU u;
  u.p.lo = *(const v4u*)(rowk + h * 8);
  u.p.hi = *(const v4u*)(rowk + 16 + h * 8);
  return u.v;
}
// B fragment (32x16 bf16, KxN): lane = column; h=0 -> K[k0..k0+7] + K[k0+16..k0+23], h=1 -> K[k0+8..k0+15] + K[k0+24..k0+31].
__device__ __forceinline__ v16bf load_frag_b(const __bf16* rowk, int h) {
  FragU u;
  u.p.lo = *(const v4u*)(rowk + h * 8);
  u.p.hi = *(const v4u*)(rowk + 16 + h * 8);
  return u.v;
}
__device__ __forceinline__ v8f wmma_bf16(v16bf a, v16bf b, v8f c) {
  return __builtin_amdgcn_wmma_f32_16x16x32_bf16(false, a, false, b, (short)0, c,
                                                 false, false);
}

// ---------------- Kernel A: transpose x (B,C,N) -> xtbf (M,C) bf16 + fp32 sq ----
__global__ __launch_bounds__(256) void k_prep(const float* __restrict__ x,
                                              __bf16* __restrict__ xtbf,
                                              float* __restrict__ sq) {
  int m = blockIdx.x * 256 + threadIdx.x;          // 0..8191
  int b = m >> 12, n = m & (NP - 1);
  const float* xp = x + (size_t)b * CCH * NP + n;  // stride NP over channels
  __bf16* op = xtbf + (size_t)m * CCH;
  float acc = 0.f;
  for (int c0 = 0; c0 < CCH; c0 += 8) {
    Pack8 pk;
#pragma unroll
    for (int q = 0; q < 8; ++q) {
      float v = xp[(size_t)(c0 + q) * NP];
      acc += v * v;
      pk.h[q] = (__bf16)v;
    }
    *(v4u*)(op + c0) = pk.u;                       // 16B packed store
  }
  sq[m] = acc;
}

// ---------------- Kernel W: W fp32 (C,2C) -> bf16 --------------------------------
__global__ __launch_bounds__(256) void k_wcvt(const float* __restrict__ W,
                                              __bf16* __restrict__ wbf) {
  int e = blockIdx.x * 256 + threadIdx.x;
  if (e < CCH * 2 * CCH) wbf[e] = (__bf16)W[e];
}

// ---------------- Kernel B1: gram via WMMA + fused partial top-K -----------------
// One wave per (16-query tile, j-split); grid = MTOT/16 * JSPLIT = 4096 single-wave
// workgroups (~4 waves/SIMD on a big WGP array). Emits a sorted top-16 per row
// for its 1/8th of the candidate range.
__global__ __launch_bounds__(32) void k_knn_part(const __bf16* __restrict__ xtbf,
                                                 const float* __restrict__ sq,
                                                 float* __restrict__ pls,
                                                 int* __restrict__ pli) {
  __shared__ float s_dist[16 * 17];   // padded rows: no LDS bank conflicts
  __shared__ float s_ms[512];
  __shared__ int   s_mi[512];

  int lane = threadIdx.x;
  int col = lane & 15, h = lane >> 4;
  int blk = blockIdx.x;
  int spl = blk & (JSPLIT - 1);
  int t = blk >> 3;                   // 512 query tiles
  int b = t >> 8;                     // 256 i-tiles per batch
  int it = t & 255;
  int base = b * NP;                  // global point base of this batch

  // Preload all A fragments for this 16-row query tile (full C = 8 k-steps).
  const __bf16* abase = xtbf + (size_t)(base + it * 16 + col) * CCH;
  v16bf aF[8];
#pragma unroll
  for (int ks = 0; ks < 8; ++ks) aF[ks] = load_frag_a(abase + ks * 32, h);

  // Per-lane sorted top-16 list over its parity (h) of candidate columns.
  float best[KNN];
  int bj[KNN];
#pragma unroll
  for (int q = 0; q < KNN; ++q) { best[q] = 3.0e38f; bj[q] = base; }

  int jt0 = spl * JT_PER;
  for (int jt = jt0; jt < jt0 + JT_PER; ++jt) {
    int j0 = jt * 16;
    const __bf16* bbase = xtbf + (size_t)(base + j0 + col) * CCH;
    if (jt + 1 < jt0 + JT_PER)        // warm rows j0+1..j0+16 (global_prefetch_b8)
      __builtin_prefetch(bbase + CCH + (size_t)h * 128, 0, 3);
    v8f acc = {};
#pragma unroll
    for (int ks = 0; ks < 8; ++ks) {
      v16bf bF = load_frag_b(bbase + ks * 32, h);
      acc = wmma_bf16(aF[ks], bF, acc);           // gram[i, j] chained accumulate
    }
    float sqj = sq[base + j0 + col];
    // score = sq_j - 2*gram (sq_i constant per row -> rank-equivalent to dist)
#pragma unroll
    for (int r = 0; r < 8; ++r)
      s_dist[(r + h * 8) * 17 + col] = __builtin_fmaf(-2.f, acc[r], sqj);
    __syncthreads();                  // single-wave WG: lowers to s_wait_dscnt only

    int row = col;                    // lane's query row; parity h of columns
#pragma unroll
    for (int q8 = 0; q8 < 8; ++q8) {
      int c2 = q8 * 2 + h;
      float s = s_dist[row * 17 + c2];
      if (s < best[KNN - 1]) {        // early-exit guard (common case)
        float cs = s;
        int cj = base + j0 + c2;
#pragma unroll
        for (int q = 0; q < KNN; ++q) {  // register insertion sort
          if (cs < best[q]) {
            float tf = best[q]; int ti = bj[q];
            best[q] = cs; bj[q] = cj; cs = tf; cj = ti;
          }
        }
      }
    }
    __syncthreads();
  }

  // Merge even/odd parity lists per row -> one sorted 16-list per row per split.
#pragma unroll
  for (int q = 0; q < KNN; ++q) {
    s_ms[lane * 16 + q] = best[q];
    s_mi[lane * 16 + q] = bj[q];
  }
  __syncthreads();
  if (lane < 16) {
    int ia = lane * 16, ea = ia + 16;
    int ib = (lane + 16) * 16, eb = ib + 16;
    int mi = base + it * 16 + lane;
    size_t ob = ((size_t)mi * JSPLIT + spl) * KNN;
    for (int q = 0; q < KNN; ++q) {
      float s; int j;
      bool takeA = (ib >= eb) || ((ia < ea) && (s_ms[ia] <= s_ms[ib]));
      if (takeA) { s = s_ms[ia]; j = s_mi[ia]; ++ia; }
      else       { s = s_ms[ib]; j = s_mi[ib]; ++ib; }
      pls[ob + q] = s;
      pli[ob + q] = j;
    }
  }
}

// ---------------- Kernel B2: merge partial lists + cosine weights ----------------
// One thread per query row; sublists are sorted ascending, so break out of a
// sublist as soon as its head can't beat the current 16th-best.
__global__ __launch_bounds__(128) void k_knn_merge(const float* __restrict__ pls,
                                                   const int* __restrict__ pli,
                                                   const float* __restrict__ sq,
                                                   int* __restrict__ idxg,
                                                   float* __restrict__ wtg) {
  int m = blockIdx.x * 128 + threadIdx.x;   // 0..8191
  float best[KNN];
  int bj[KNN];
#pragma unroll
  for (int q = 0; q < KNN; ++q) { best[q] = 3.0e38f; bj[q] = 0; }
  const float* ps = pls + (size_t)m * JSPLIT * KNN;
  const int* pi = pli + (size_t)m * JSPLIT * KNN;
  for (int s = 0; s < JSPLIT; ++s) {
    for (int e = 0; e < KNN; ++e) {
      float cs = ps[s * KNN + e];
      if (cs >= best[KNN - 1]) break;       // rest of this sorted sublist loses too
      int cj = pi[s * KNN + e];
#pragma unroll
      for (int q = 0; q < KNN; ++q) {
        if (cs < best[q]) {
          float tf = best[q]; int ti = bj[q];
          best[q] = cs; bj[q] = cj; cs = tf; cj = ti;
        }
      }
    }
  }
  float sqi = sq[m];
#pragma unroll
  for (int q = 0; q < KNN; ++q) {
    float s = best[q];
    int j = bj[q];
    float sqj = sq[j];
    float g = 0.5f * (sqj - s);             // recover gram from score
    wtg[m * KNN + q] = g * rsqrtf(sqi * sqj);
    idxg[m * KNN + q] = j;
  }
}

// ---------------- Kernel C: y1 = X*W1^T, y2 = X*W2^T via WMMA --------------------
// One wave per 16x16 output tile; computes both halves of W (16 WMMAs / wave).
__global__ __launch_bounds__(128) void k_gemm(const __bf16* __restrict__ xtbf,
                                              const __bf16* __restrict__ wbf,
                                              float* __restrict__ y1,
                                              float* __restrict__ y2) {
  int wave = blockIdx.x * (blockDim.x >> 5) + (threadIdx.x >> 5);
  int lane = threadIdx.x & 31;
  int col = lane & 15, h = lane >> 4;
  int mt = wave >> 4;                  // 512 m-tiles
  int ct = wave & 15;                  // 16 c-tiles
  const __bf16* abase = xtbf + (size_t)(mt * 16 + col) * CCH;
  const __bf16* bbase = wbf + (size_t)(ct * 16 + col) * (2 * CCH);
  v8f a1 = {}, a2 = {};
#pragma unroll
  for (int ks = 0; ks < 8; ++ks) {
    v16bf a  = load_frag_a(abase + ks * 32, h);
    v16bf b1 = load_frag_b(bbase + ks * 32, h);         // W[:, :C]
    v16bf b2 = load_frag_b(bbase + CCH + ks * 32, h);   // W[:, C:]
    a1 = wmma_bf16(a, b1, a1);
    a2 = wmma_bf16(a, b2, a2);
  }
  int c = ct * 16 + col;
#pragma unroll
  for (int r = 0; r < 8; ++r) {
    int m = mt * 16 + r + h * 8;
    y1[(size_t)m * CCH + c] = a1[r];
    y2[(size_t)m * CCH + c] = a2[r];
  }
}

// ---------------- Kernel D: gather + weight + bias + max over K ------------------
__global__ __launch_bounds__(256) void k_out(const float* __restrict__ y1,
                                             const float* __restrict__ y2,
                                             const int* __restrict__ idxg,
                                             const float* __restrict__ wtg,
                                             const float* __restrict__ bias,
                                             float* __restrict__ out) {
  int t = threadIdx.x;                 // channel c
  int g = blockIdx.x;                  // 32-point group (256 groups)
  int m0 = g * 32;
  int b = m0 >> 12, n0 = m0 & (NP - 1);
  float bv = bias[t];
  float* obase = out + (size_t)b * CCH * NP + (size_t)t * NP + n0;  // contiguous in n
  for (int p4 = 0; p4 < 32; p4 += 4) {
    float tmp[4];
#pragma unroll
    for (int pp = 0; pp < 4; ++pp) {
      int m = m0 + p4 + pp;
      float ctr = y2[(size_t)m * CCH + t] + bv;
      float acc = -3.0e38f;
#pragma unroll
      for (int k = 0; k < KNN; ++k) {
        int j = idxg[m * KNN + k];
        float w = wtg[m * KNN + k];
        acc = fmaxf(acc, w * (y1[(size_t)j * CCH + t] + ctr));
      }
      tmp[pp] = acc;
    }
    *(float4*)(obase + p4) = make_float4(tmp[0], tmp[1], tmp[2], tmp[3]);
  }
}

extern "C" void kernel_launch(void* const* d_in, const int* in_sizes, int n_in,
                              void* d_out, int out_size, void* d_ws, size_t ws_size,
                              hipStream_t stream) {
  const float* x    = (const float*)d_in[0];   // (B, C, N)
  const float* W    = (const float*)d_in[1];   // (C, 2C)
  const float* bias = (const float*)d_in[2];   // (C,)
  float* out = (float*)d_out;

  char* ws = (char*)d_ws;
  size_t o = 0;
  auto alignup = [](size_t v) { return (v + 255) & ~(size_t)255; };
  __bf16* xtbf = (__bf16*)(ws + o); o = alignup(o + (size_t)MTOT * CCH * 2);
  __bf16* wbf  = (__bf16*)(ws + o); o = alignup(o + (size_t)CCH * 2 * CCH * 2);
  float*  sq   = (float*)(ws + o);  o = alignup(o + (size_t)MTOT * 4);
  int*    idxg = (int*)(ws + o);    o = alignup(o + (size_t)MTOT * KNN * 4);
  float*  wtg  = (float*)(ws + o);  o = alignup(o + (size_t)MTOT * KNN * 4);
  float*  y1   = (float*)(ws + o);  o = alignup(o + (size_t)MTOT * CCH * 4);
  float*  y2   = (float*)(ws + o);  o = alignup(o + (size_t)MTOT * CCH * 4);
  float*  pls  = (float*)(ws + o);  o = alignup(o + (size_t)MTOT * JSPLIT * KNN * 4);
  int*    pli  = (int*)(ws + o);    o = alignup(o + (size_t)MTOT * JSPLIT * KNN * 4);
  (void)ws_size; (void)in_sizes; (void)n_in; (void)out_size;

  k_prep<<<MTOT / 256, 256, 0, stream>>>(x, xtbf, sq);
  k_wcvt<<<(CCH * 2 * CCH + 255) / 256, 256, 0, stream>>>(W, wbf);
  k_knn_part<<<(MTOT / 16) * JSPLIT, 32, 0, stream>>>(xtbf, sq, pls, pli);
  k_knn_merge<<<MTOT / 128, 128, 0, stream>>>(pls, pli, sq, idxg, wtg);
  k_gemm<<<(MTOT / 16) * (CCH / 16) / 4, 128, 0, stream>>>(xtbf, wbf, y1, y2);
  k_out<<<MTOT / 32, 256, 0, stream>>>(y1, y2, idxg, wtg, bias, out);
}
